// DistributeTriangleMultiplication_49460843380748
// MI455X (gfx1250) — matrix operand order repaired and checked by reading, not compiled, not run on hardware
//
#include <hip/hip_runtime.h>

typedef __attribute__((ext_vector_type(16))) __bf16 v16bf;
typedef __attribute__((ext_vector_type(8)))  __bf16 v8bf;
typedef __attribute__((ext_vector_type(8)))  float  v8f;
typedef __attribute__((ext_vector_type(4)))  unsigned su4;
typedef __attribute__((ext_vector_type(8)))  unsigned su8;

#define NN 589824L   // 768*768
#define NDIM 768
#define CDIM 128

__device__ __forceinline__ unsigned short f2bf(float f) {
    unsigned u = __float_as_uint(f);
    unsigned r = u + 0x7FFFu + ((u >> 16) & 1u);
    return (unsigned short)(r >> 16);
}

__device__ __forceinline__ float sigmoidf(float x) {
    return 1.0f / (1.0f + __expf(-x));
}

__device__ __forceinline__ float wave_sum(float v) {
    #pragma unroll
    for (int off = 16; off > 0; off >>= 1) v += __shfl_xor(v, off, 32);
    return v;
}

// A-fragment (16-bit, 16x32): lane (m = L%16, hi = L/16) holds
// k = {base..base+7, base+16..base+23} with base = k0 + hi*8.
__device__ __forceinline__ v16bf frag_a(const unsigned short* p) {
    v8bf lo = *(const v8bf*)p;
    v8bf hi = *(const v8bf*)(p + 16);
    return __builtin_shufflevector(lo, hi, 0,1,2,3,4,5,6,7,8,9,10,11,12,13,14,15);
}

// B-fragment (16-bit, 32x16): lane (n = L%16, hi = L/16) holds column n,
// k = k0 + hi*16 .. +15 contiguous.
__device__ __forceinline__ v16bf frag_b16(const unsigned short* p) {
    v8bf lo = *(const v8bf*)p;
    v8bf hi = *(const v8bf*)(p + 8);
    return __builtin_shufflevector(lo, hi, 0,1,2,3,4,5,6,7,8,9,10,11,12,13,14,15);
}

__device__ __forceinline__ v8f wmma_bf16(v16bf a, v16bf b, v8f c) {
    return __builtin_amdgcn_wmma_f32_16x16x32_bf16(false, a, false, b, (short)0, c, false, false);
}

// TDM: DMA a 128-row x 32-element (bf16) 2D tile, row stride 768 elements,
// into LDS with 16B padding after each 64B row (-> 40-short LDS row stride).
// D# groups 0 and 1 only (2D tensor, VADDR2/VADDR3 = NULL form).
__device__ __forceinline__ void tdm_load_tile(unsigned lds_off, const unsigned short* gptr) {
    unsigned long long ga = (unsigned long long)gptr;
    su4 g0;
    g0[0] = 1u;                                   // count=1, user descriptor
    g0[1] = lds_off;                              // lds_addr (bytes)
    g0[2] = (unsigned)ga;                         // global_addr[31:0]
    g0[3] = (unsigned)(ga >> 32) | 0x80000000u;   // global_addr[56:32] | type=2
    su8 g1;
    g1[0] = (1u << 16)      // data_size = 2 bytes
          | (1u << 20)      // pad_enable
          | (3u << 22)      // pad_interval: 16 DWORDs (64B)
          | (3u << 25);     // pad_amount: 4 DWORDs (16B)
    g1[1] = 768u << 16;     // tensor_dim0[15:0] in [31:16]
    g1[2] = 768u << 16;     // tensor_dim0[31:16]=0 | tensor_dim1[15:0] in [31:16]
    g1[3] = 32u << 16;      // tensor_dim1[31:16]=0 | tile_dim0=32 in [31:16]
    g1[4] = 128u;           // tile_dim1=128, tile_dim2=0
    g1[5] = 768u;           // tensor_dim0_stride[31:0]
    g1[6] = 0u;             // stride high bits / tensor_dim1_stride low
    g1[7] = 0u;
    asm volatile("tensor_load_to_lds %0, %1" :: "s"(g0), "s"(g1) : "memory");
}

// ---------------- K0: weight fp32 -> bf16 ----------------
// wb layout: proj[0,32768) gate[32768,65536) out_w[65536,81920) gating[81920,98304)
__global__ void cvt_weights(const float* __restrict__ pw, const float* __restrict__ gw,
                            const float* __restrict__ ow, const float* __restrict__ qw,
                            unsigned short* __restrict__ wb) {
    int i = blockIdx.x * 256 + threadIdx.x;
    float v;
    if      (i < 32768) v = pw[i];
    else if (i < 65536) v = gw[i - 32768];
    else if (i < 81920) v = ow[i - 65536];
    else                v = qw[i - 81920];
    wb[i] = f2bf(v);
}

// ---------------- K1: LN1, pair fp32 -> x bf16 (NN, C) ----------------
__global__ void __launch_bounds__(256) ln1_kernel(const float* __restrict__ pair,
                                                  const float* __restrict__ w,
                                                  const float* __restrict__ b,
                                                  unsigned short* __restrict__ xb) {
    int lane = threadIdx.x & 31;
    int wv   = threadIdx.x >> 5;
    long row = (long)blockIdx.x * 8 + wv;
    float4 v = *(const float4*)(pair + row * CDIM + lane * 4);
    float mean = wave_sum(v.x + v.y + v.z + v.w) * (1.0f / 128.0f);
    float dx = v.x - mean, dy = v.y - mean, dz = v.z - mean, dw = v.w - mean;
    float var = wave_sum(dx*dx + dy*dy + dz*dz + dw*dw) * (1.0f / 128.0f);
    float rs = rsqrtf(var + 1e-5f);
    float4 w4 = *(const float4*)(w + lane * 4);
    float4 b4 = *(const float4*)(b + lane * 4);
    unsigned u0 = (unsigned)f2bf(dx*rs*w4.x + b4.x) | ((unsigned)f2bf(dy*rs*w4.y + b4.y) << 16);
    unsigned u1 = (unsigned)f2bf(dz*rs*w4.z + b4.z) | ((unsigned)f2bf(dw*rs*w4.w + b4.w) << 16);
    uint2 o; o.x = u0; o.y = u1;
    *(uint2*)(xb + row * CDIM + lane * 4) = o;
}

// ---------------- K2: proj+gate WMMA, mask*sigmoid, scatter to a/b ----------------
// D = W (16ch x 32k) x X^T (32k x 16pos). grid (NN/16, 2), 8 waves/block.
__global__ void __launch_bounds__(256) proj_gate_kernel(const unsigned short* __restrict__ xb,
                                                        const unsigned short* __restrict__ wb,
                                                        const float* __restrict__ mask,
                                                        unsigned short* __restrict__ ab,
                                                        unsigned short* __restrict__ bb) {
    int lane = threadIdx.x & 31;
    int wv   = threadIdx.x >> 5;
    int lrow = lane & 15;
    int hi   = lane >> 4;
    long m0  = (long)blockIdx.x * 16;
    int  c0  = blockIdx.y * 128 + wv * 16;
    const unsigned short* xrow = xb + (m0 + lrow) * CDIM;
    const unsigned short* wp   = wb + (c0 + lrow) * CDIM;
    const unsigned short* wg   = wb + 32768 + (c0 + lrow) * CDIM;
    v8f accP = {}; v8f accG = {};
    #pragma unroll
    for (int kk = 0; kk < 4; ++kk) {
        int ka = kk * 32 + hi * 8;
        int kb = kk * 32 + hi * 16;
        v16bf bx = frag_a(xrow + ka);
        v16bf ap = frag_a(wp + ka);
        v16bf ag = frag_a(wg + ka);
        accP = wmma_bf16(ap, bx, accP);
        accG = wmma_bf16(ag, bx, accG);
    }
    long m = m0 + lrow;
    float mk = mask[m];
    #pragma unroll
    for (int r = 0; r < 8; ++r) {
        int ch = c0 + r + hi * 8;                    // D: M = r + 8*hi, N = lrow
        float val = accP[r] * mk * sigmoidf(accG[r]);
        unsigned short* tgt = (ch & 1) ? bb : ab;
        tgt[(long)(ch >> 1) * NN + m] = f2bf(val);
    }
}

// ---------------- K3: per-channel out[c] = a_c @ b_c^T, 128x128 tile ----------------
// TDM double-buffered: wave 0 DMAs next K-tile while all 8 waves run WMMA.
__global__ void __launch_bounds__(256) einsum_kernel(const unsigned short* __restrict__ ab,
                                                     const unsigned short* __restrict__ bb,
                                                     float* __restrict__ e) {
    __shared__ __align__(16) unsigned short As[2][128][40];   // 40-short rows via TDM pad
    __shared__ __align__(16) unsigned short Bs[2][128][40];
    int c = blockIdx.z;
    int i0 = blockIdx.x * 128, j0 = blockIdx.y * 128;
    const unsigned short* ap = ab + (long)c * NN + (long)i0 * NDIM;
    const unsigned short* bp = bb + (long)c * NN + (long)j0 * NDIM;
    int t = threadIdx.x;
    int lane = t & 31, wv = t >> 5;
    int lrow = lane & 15, hi = lane >> 4;
    unsigned ldsA[2], ldsB[2];
    ldsA[0] = (unsigned)(unsigned long long)(&As[0][0][0]);
    ldsA[1] = (unsigned)(unsigned long long)(&As[1][0][0]);
    ldsB[0] = (unsigned)(unsigned long long)(&Bs[0][0][0]);
    ldsB[1] = (unsigned)(unsigned long long)(&Bs[1][0][0]);
    v8f acc[8];
    #pragma unroll
    for (int q = 0; q < 8; ++q) acc[q] = (v8f){};
    if (wv == 0) {
        tdm_load_tile(ldsA[0], ap);
        tdm_load_tile(ldsB[0], bp);
    }
    for (int s = 0; s < 24; ++s) {
        int cur = s & 1;
        if (wv == 0) {
            if (s < 23) {
                tdm_load_tile(ldsA[cur ^ 1], ap + (s + 1) * 32);
                tdm_load_tile(ldsB[cur ^ 1], bp + (s + 1) * 32);
                __builtin_amdgcn_s_wait_tensorcnt(2);  // pair for step s complete
            } else {
                __builtin_amdgcn_s_wait_tensorcnt(0);
            }
        }
        __syncthreads();                               // tile[cur] visible to all waves
        v16bf af = frag_a(&As[cur][wv * 16 + lrow][hi * 8]);
        #pragma unroll
        for (int ct = 0; ct < 8; ++ct) {
            v16bf bv = frag_a(&Bs[cur][ct * 16 + lrow][hi * 8]);
            acc[ct] = wmma_bf16(af, bv, acc[ct]);
        }
        __syncthreads();                               // done reading before DMA overwrites
    }
    float* ep = e + (long)c * NN;
    #pragma unroll
    for (int ct = 0; ct < 8; ++ct) {
        #pragma unroll
        for (int r = 0; r < 8; ++r) {
            int i = i0 + wv * 16 + r + hi * 8;
            int j = j0 + ct * 16 + lrow;
            ep[(long)i * NDIM + j] = acc[ct][r];
        }
    }
}

// ---------------- K4: transpose + LN2 + (x out_w^T)*sigmoid(x gating_w^T) ----------------
// 32 positions per block, 8 waves. grid NN/32.
__global__ void __launch_bounds__(256) final_kernel(const float* __restrict__ e,
                                                    const unsigned short* __restrict__ xb,
                                                    const unsigned short* __restrict__ w_out,
                                                    const unsigned short* __restrict__ w_gate,
                                                    const float* __restrict__ ln2w,
                                                    const float* __restrict__ ln2b,
                                                    float* __restrict__ out) {
    __shared__ __align__(16) float          et[32][132];
    __shared__ __align__(16) unsigned short lnb[32][136];
    __shared__ __align__(16) unsigned short xgb[32][136];
    long m0 = (long)blockIdx.x * 32;
    int t = threadIdx.x;
    int lane = t & 31, wv = t >> 5;
    // gather e (C,NN) -> et[m][c], coalesced per channel
    #pragma unroll
    for (int it = 0; it < 16; ++it) {
        int idx = t + it * 256;               // 4096 floats
        int c = idx >> 5, mm = idx & 31;
        et[mm][c] = e[(long)c * NN + m0 + mm];
    }
    // x rows (already bf16, row-major)
    #pragma unroll
    for (int it = 0; it < 2; ++it) {
        int idx = t + it * 256;               // 512 segments of 8 bf16
        int mm = idx >> 4, seg = idx & 15;
        *(uint4*)(&xgb[mm][seg * 8]) = *(const uint4*)(xb + (m0 + mm) * CDIM + seg * 8);
    }
    __syncthreads();
    float4 w4 = *(const float4*)(ln2w + lane * 4);
    float4 b4 = *(const float4*)(ln2b + lane * 4);
    #pragma unroll
    for (int rr = 0; rr < 4; ++rr) {
        int mm = rr * 8 + wv;
        float4 v = *(const float4*)(&et[mm][lane * 4]);
        float mean = wave_sum(v.x + v.y + v.z + v.w) * (1.0f / 128.0f);
        float dx = v.x - mean, dy = v.y - mean, dz = v.z - mean, dw = v.w - mean;
        float var = wave_sum(dx*dx + dy*dy + dz*dz + dw*dw) * (1.0f / 128.0f);
        float rs = rsqrtf(var + 1e-5f);
        unsigned u0 = (unsigned)f2bf(dx*rs*w4.x + b4.x) | ((unsigned)f2bf(dy*rs*w4.y + b4.y) << 16);
        unsigned u1 = (unsigned)f2bf(dz*rs*w4.z + b4.z) | ((unsigned)f2bf(dw*rs*w4.w + b4.w) << 16);
        uint2 o; o.x = u0; o.y = u1;
        *(uint2*)(&lnb[mm][lane * 4]) = o;
    }
    __syncthreads();
    int strip = wv & 1;                // 2 row strips of 16
    int cb    = (wv >> 1) * 32;        // 4 groups x 32 channels
    int lrow  = lane & 15, hi = lane >> 4;
    int arow  = strip * 16 + lrow;
    v8f accO[2], accG[2];
    #pragma unroll
    for (int q = 0; q < 2; ++q) { accO[q] = (v8f){}; accG[q] = (v8f){}; }
    #pragma unroll
    for (int kk = 0; kk < 4; ++kk) {
        int ka = kk * 32 + hi * 8;
        int kb = kk * 32 + hi * 16;
        v16bf aln = frag_a(&lnb[arow][ka]);
        v16bf axg = frag_a(&xgb[arow][ka]);
        #pragma unroll
        for (int ct = 0; ct < 2; ++ct) {
            int cc = cb + ct * 16 + lrow;
            v16bf bo = frag_a(w_out  + cc * CDIM + ka);
            v16bf bg = frag_a(w_gate + cc * CDIM + ka);
            accO[ct] = wmma_bf16(aln, bo, accO[ct]);
            accG[ct] = wmma_bf16(axg, bg, accG[ct]);
        }
    }
    #pragma unroll
    for (int ct = 0; ct < 2; ++ct) {
        #pragma unroll
        for (int r = 0; r < 8; ++r) {
            long m = m0 + strip * 16 + r + hi * 8;
            int cc = cb + ct * 16 + lrow;
            out[m * CDIM + cc] = accO[ct][r] * sigmoidf(accG[ct][r]);
        }
    }
}

extern "C" void kernel_launch(void* const* d_in, const int* in_sizes, int n_in,
                              void* d_out, int out_size, void* d_ws, size_t ws_size,
                              hipStream_t stream) {
    const float* pair     = (const float*)d_in[0];
    const float* mask     = (const float*)d_in[1];
    const float* ln1_w    = (const float*)d_in[2];
    const float* ln1_b    = (const float*)d_in[3];
    const float* proj_w   = (const float*)d_in[4];
    const float* gate_w   = (const float*)d_in[5];
    const float* ln2_w    = (const float*)d_in[6];
    const float* ln2_b    = (const float*)d_in[7];
    const float* out_w    = (const float*)d_in[8];
    const float* gating_w = (const float*)d_in[9];
    float* out = (float*)d_out;

    char* ws = (char*)d_ws;
    unsigned short* xb = (unsigned short*)(ws);                 // 151 MB  x bf16 (NN,C)
    unsigned short* ab = (unsigned short*)(ws + 150994944L);    // 151 MB  a bf16 (C,N,N)
    unsigned short* bb = (unsigned short*)(ws + 301989888L);    // 151 MB  b bf16 (C,N,N)
    float*          e  = (float*)         (ws + 452984832L);    // 302 MB  einsum fp32 (C,N,N)
    unsigned short* wb = (unsigned short*)(ws + 754974720L);    // 192 KB  weights bf16

    cvt_weights<<<384, 256, 0, stream>>>(proj_w, gate_w, out_w, gating_w, wb);
    ln1_kernel<<<73728, 256, 0, stream>>>(pair, ln1_w, ln1_b, xb);
    proj_gate_kernel<<<dim3(36864, 2), 256, 0, stream>>>(xb, wb, mask, ab, bb);
    einsum_kernel<<<dim3(6, 6, 128), 256, 0, stream>>>(ab, bb, e);
    final_kernel<<<18432, 256, 0, stream>>>(e, xb, wb + 65536, wb + 81920, ln2_w, ln2_b, out);
}
